// ObligationEmbedding_83365315215756
// MI455X (gfx1250) — matrix operand order repaired and checked by reading, not compiled, not run on hardware
//
#include <hip/hip_runtime.h>
#include <hip/hip_bf16.h>
#include <math.h>
#include <stdint.h>

typedef __attribute__((ext_vector_type(16))) __bf16 v16bf;
typedef __attribute__((ext_vector_type(8)))  __bf16 v8bf;
typedef __attribute__((ext_vector_type(8)))  float  v8f;
typedef int v4i __attribute__((vector_size(16)));

#define HID   500
#define HP    512
#define G4P   2048   // padded 4*H
#define SEQ   128
#define NB    256    // goals batch
#define NHYP  2048
#define NTOT  2304
#define NVOC  1000
#define ROWS_PER_WG 32

#if __has_builtin(__builtin_amdgcn_global_load_async_to_lds_b128) && \
    __has_builtin(__builtin_amdgcn_s_wait_asynccnt)
#define USE_ASYNC_LDS 1
#else
#define USE_ASYNC_LDS 0
#endif

__device__ __forceinline__ float fast_rcp(float x) {
#if __has_builtin(__builtin_amdgcn_rcpf)
  return __builtin_amdgcn_rcpf(x);
#else
  return 1.0f / x;
#endif
}
// sigmoid(x) = rcp(1 + e^-x) ; tanh(x) = 1 - 2*rcp(e^{2x}+1)  (correct at +-inf)
__device__ __forceinline__ float sigm(float x) { return fast_rcp(1.0f + __expf(-x)); }
__device__ __forceinline__ float tanh_fast(float x) {
  return 1.0f - 2.0f * fast_rcp(__expf(2.0f * x) + 1.0f);
}

// ---------------- prepack kernels ----------------
__global__ void pack_emb_k(const float* __restrict__ emb, __bf16* __restrict__ emb_p) {
  int idx = blockIdx.x * 256 + threadIdx.x;
  if (idx >= NVOC * HP) return;
  int r = idx >> 9, k = idx & 511;
  float v = (k < HID) ? emb[r * HID + k] : 0.0f;
  emb_p[idx] = (__bf16)v;
}

__global__ void pack_w_k(const float* __restrict__ W, __bf16* __restrict__ Wp) {
  int idx = blockIdx.x * 256 + threadIdx.x;
  if (idx >= G4P * HP) return;
  int j = idx >> 9, k = idx & 511;
  int q = j >> 9, jj = j & 511;
  float v = (jj < HID && k < HID) ? W[(q * HID + jj) * HID + k] : 0.0f;
  Wp[idx] = (__bf16)v;
}

__global__ void pack_bias_k(const float* __restrict__ bih, const float* __restrict__ bhh,
                            float* __restrict__ bp) {
  int j = blockIdx.x * 256 + threadIdx.x;
  if (j >= G4P) return;
  int q = j >> 9, jj = j & 511;
  bp[j] = (jj < HID) ? (bih[q * HID + jj] + bhh[q * HID + jj]) : 0.0f;
}

__global__ void zero_f32_k(float* __restrict__ p, int n) {
  int i = blockIdx.x * 256 + threadIdx.x;
  if (i < n) p[i] = 0.0f;
}

// ---------------- LSTM kernel (WMMA bf16, 32 rows / WG) ----------------
// One WG = 32 rows (2 WMMA M-tiles). 512 threads = 16 waves.
// Each wave owns 2 u-groups x 4 gates; every B fragment (weight tile from L2)
// is reused for BOTH M-tiles -> halves L2 weight bandwidth vs 16-row tiling.
// x_t staging is double-buffered: the async-to-LDS gather for step t+1 is
// issued before the WMMA phase of step t, overlapping copy with compute.
// c lives in registers, h in LDS bf16.
__global__ __launch_bounds__(512, 1)
void lstm_wmma_k(const int* __restrict__ goals, const int* __restrict__ hyps,
                 const __bf16* __restrict__ emb_p,
                 const __bf16* __restrict__ Wih_p, const __bf16* __restrict__ Whh_p,
                 const float* __restrict__ bias_p, float* __restrict__ Hall) {
  __shared__ __bf16 xbuf[2][ROWS_PER_WG * HP];   // 2 x 32 KB
  __shared__ __bf16 hbuf[ROWS_PER_WG * HP];      // 32 KB

  const int tid    = threadIdx.x;
  const int lane   = tid & 31;
  const int wv     = tid >> 5;       // 0..15
  const int lm     = lane & 15;      // n index / A-matrix M (within tile)
  const int laneHi = lane >> 4;      // 0/1 half-wave
  const int rowbase = blockIdx.x * ROWS_PER_WG;

  const int xm = tid >> 4;          // staging row (16 threads per row, 32 rows)
  const int xc = (tid & 15) * 32;   // staging column base (32 bf16 = 64 B per thread)
  const int xrow = rowbase + xm;
  const int* xtokens = (xrow < NB) ? (goals + xrow * SEQ) : (hyps + (xrow - NB) * SEQ);

  auto stage_x = [&](int t, int bi) {
    int tok = xtokens[t];
    const __bf16* src = emb_p + tok * HP + xc;
    __bf16* dst = &xbuf[bi][xm * HP + xc];
#if USE_ASYNC_LDS
    __builtin_amdgcn_global_load_async_to_lds_b128((v4i*)src, (v4i*)dst, 0, 0);
    __builtin_amdgcn_global_load_async_to_lds_b128((v4i*)src, (v4i*)dst, 16, 0);
    __builtin_amdgcn_global_load_async_to_lds_b128((v4i*)src, (v4i*)dst, 32, 0);
    __builtin_amdgcn_global_load_async_to_lds_b128((v4i*)src, (v4i*)dst, 48, 0);
#else
    *(v16bf*)(dst)      = *(const v16bf*)(src);
    *(v16bf*)(dst + 16) = *(const v16bf*)(src + 16);
#endif
  };

  // init h = 0
  for (int i = tid; i < ROWS_PER_WG * HP; i += 512) hbuf[i] = (__bf16)0.0f;

  float creg[2][2][8];               // [group][mtile][vgpr]
#pragma unroll
  for (int gi = 0; gi < 2; ++gi)
#pragma unroll
    for (int mt = 0; mt < 2; ++mt)
#pragma unroll
      for (int v = 0; v < 8; ++v) creg[gi][mt][v] = 0.0f;

  float bias_r[2][4];
#pragma unroll
  for (int gi = 0; gi < 2; ++gi) {
    int ub = 16 * (2 * wv + gi) + lm;
#pragma unroll
    for (int q = 0; q < 4; ++q) bias_r[gi][q] = bias_p[q * HP + ub];
  }

  // prologue: stage x_0
  stage_x(0, 0);
#if USE_ASYNC_LDS
  __builtin_amdgcn_s_wait_asynccnt(0);
#endif
  __syncthreads();

  for (int t = 0; t < SEQ; ++t) {
    // overlap: kick off gather of x_{t+1} into the other buffer while we do WMMAs
    if (t + 1 < SEQ) stage_x(t + 1, (t + 1) & 1);

    v8f acc[2][4][2];                // [group][gate][mtile]
#pragma unroll
    for (int gi = 0; gi < 2; ++gi)
#pragma unroll
      for (int q = 0; q < 4; ++q)
#pragma unroll
        for (int mt = 0; mt < 2; ++mt)
#pragma unroll
          for (int v = 0; v < 8; ++v) acc[gi][q][mt][v] = bias_r[gi][q];

    const __bf16* xcur = xbuf[t & 1];
#pragma unroll
    for (int half = 0; half < 2; ++half) {
      const __bf16* abase = (half == 0) ? xcur : hbuf;       // both LDS
      const __bf16* Wp    = (half == 0) ? Wih_p : Whh_p;
#pragma unroll 1
      for (int kb = 0; kb < 16; ++kb) {
        // A fragments for both M-tiles:
        // lane (lm, laneHi): e<8 -> K = kb*32 + laneHi*8 + e
        //                    e>=8 -> K = kb*32 + 16 + laneHi*8 + (e-8)
        v16bf a[2];
#pragma unroll
        for (int mt = 0; mt < 2; ++mt) {
          const __bf16* pa = abase + (lm + 16 * mt) * HP + kb * 32 + laneHi * 8;
          v8bf alo = *(const v8bf*)(pa);
          v8bf ahi = *(const v8bf*)(pa + 16);
#pragma unroll
          for (int e = 0; e < 8; ++e) { a[mt][e] = alo[e]; a[mt][e + 8] = ahi[e]; }
        }
#pragma unroll
        for (int gi = 0; gi < 2; ++gi) {
          int jrow = 16 * (2 * wv + gi) + lm;   // output column -> weight row
#pragma unroll
          for (int q = 0; q < 4; ++q) {
            // B fragment: same K order as A (e<8 -> laneHi*8+e, e>=8 -> 16+laneHi*8+(e-8))
            const __bf16* pb = Wp + (q * HP + jrow) * HP + kb * 32 + laneHi * 8;
            v8bf blo = *(const v8bf*)(pb);
            v8bf bhi = *(const v8bf*)(pb + 16);
            v16bf b;
#pragma unroll
            for (int e = 0; e < 8; ++e) { b[e] = blo[e]; b[e + 8] = bhi[e]; }
            // reuse the same B for both M-tiles (halves L2 weight traffic)
            acc[gi][q][0] = __builtin_amdgcn_wmma_f32_16x16x32_bf16(
                false, a[0], false, b, (short)0, acc[gi][q][0], false, false);
            acc[gi][q][1] = __builtin_amdgcn_wmma_f32_16x16x32_bf16(
                false, a[1], false, b, (short)0, acc[gi][q][1], false, false);
          }
        }
      }
    }
    __syncthreads();  // all waves done reading old h (and xcur) before overwrite

    // elementwise LSTM cell update (in registers), write new h bf16 to LDS
#pragma unroll
    for (int gi = 0; gi < 2; ++gi) {
      int u = 16 * (2 * wv + gi) + lm;
#pragma unroll
      for (int mt = 0; mt < 2; ++mt) {
#pragma unroll
        for (int v = 0; v < 8; ++v) {
          float ig = acc[gi][0][mt][v];
          float fg = acc[gi][1][mt][v];
          float gg = acc[gi][2][mt][v];
          float og = acc[gi][3][mt][v];
          float cn = sigm(fg) * creg[gi][mt][v] + sigm(ig) * tanh_fast(gg);
          float hn = sigm(og) * tanh_fast(cn);
          creg[gi][mt][v] = cn;
          int m = v + laneHi * 8 + 16 * mt;
          hbuf[m * HP + u] = (__bf16)hn;
          if (t == SEQ - 1) Hall[(rowbase + m) * HP + u] = hn;
        }
      }
    }
#if USE_ASYNC_LDS
    __builtin_amdgcn_s_wait_asynccnt(0);   // x_{t+1} landed in LDS
#endif
    __syncthreads();  // h_{t} and x_{t+1} visible to all waves
  }
}

// ---------------- segment sum ----------------
__global__ void seg_sum_k(const float* __restrict__ Hall, const int* __restrict__ seg,
                          float* __restrict__ HS) {
  int r = blockIdx.x;              // hyp row
  int s = seg[r];
  const float* src = Hall + (NB + r) * HP;
  for (int c = threadIdx.x; c < HID; c += 256)
    atomicAdd(&HS[s * HP + c], src[c]);
}

// ---------------- fused MLP + L2 norm (one block per output row) ----------------
__global__ __launch_bounds__(256)
void mlp_k(const float* __restrict__ Hall, const float* __restrict__ HS,
           const float* __restrict__ W1, const float* __restrict__ b1,
           const float* __restrict__ W2, const float* __restrict__ b2,
           const float* __restrict__ Wf, const float* __restrict__ bfin,
           float* __restrict__ out) {
  __shared__ float xr[2 * HID];
  __shared__ float y1[HID];
  __shared__ float y2[HID];
  __shared__ float red[256];
  const int b = blockIdx.x;
  const int tid = threadIdx.x;

  for (int c = tid; c < HID; c += 256) {
    xr[c]       = Hall[b * HP + c];   // goal hidden
    xr[HID + c] = HS[b * HP + c];     // hyp segment sum
  }
  __syncthreads();

  for (int u = tid; u < HID; u += 256) {
    float acc = b1[u];
    const float* w = W1 + u * (2 * HID);
    for (int k = 0; k < 2 * HID; ++k) acc = fmaf(xr[k], w[k], acc);
    y1[u] = (acc >= 0.0f) ? acc : 0.01f * acc;
  }
  __syncthreads();

  for (int u = tid; u < HID; u += 256) {
    float acc = b2[u];
    const float* w = W2 + u * HID;
    for (int k = 0; k < HID; ++k) acc = fmaf(y1[k], w[k], acc);
    y2[u] = (acc >= 0.0f) ? acc : 0.01f * acc;
  }
  __syncthreads();

  float y3v[2]; int uu[2]; int cnt = 0;
  float ss = 0.0f;
  for (int u = tid; u < HID; u += 256) {
    float acc = bfin[u];
    const float* w = Wf + u * HID;
    for (int k = 0; k < HID; ++k) acc = fmaf(y2[k], w[k], acc);
    y3v[cnt] = acc; uu[cnt] = u; ++cnt;
    ss += acc * acc;
  }
  red[tid] = ss;
  __syncthreads();
  for (int s = 128; s > 0; s >>= 1) {
    if (tid < s) red[tid] += red[tid + s];
    __syncthreads();
  }
  float nrm = fmaxf(sqrtf(red[0]), 1e-12f);
  for (int i = 0; i < cnt; ++i) out[b * HID + uu[i]] = y3v[i] / nrm;
}

// ---------------- launch ----------------
extern "C" void kernel_launch(void* const* d_in, const int* in_sizes, int n_in,
                              void* d_out, int out_size, void* d_ws, size_t ws_size,
                              hipStream_t stream) {
  (void)in_sizes; (void)n_in; (void)out_size; (void)ws_size;
  const int*   goals = (const int*)d_in[0];
  const int*   hyps  = (const int*)d_in[1];
  const int*   seg   = (const int*)d_in[2];
  const float* emb   = (const float*)d_in[3];
  const float* Wih   = (const float*)d_in[4];
  const float* Whh   = (const float*)d_in[5];
  const float* bih   = (const float*)d_in[6];
  const float* bhh   = (const float*)d_in[7];
  const float* W1    = (const float*)d_in[8];
  const float* b1    = (const float*)d_in[9];
  const float* W2    = (const float*)d_in[10];
  const float* b2    = (const float*)d_in[11];
  const float* Wf    = (const float*)d_in[12];
  const float* bfin  = (const float*)d_in[13];
  float* out = (float*)d_out;

  char* ws = (char*)d_ws;
  __bf16* emb_p  = (__bf16*)(ws + 0);          // 1000*512*2  = 1,024,000 B
  __bf16* Wih_p  = (__bf16*)(ws + 1048576);    // 2048*512*2  = 2,097,152 B
  __bf16* Whh_p  = (__bf16*)(ws + 3145728);    // 2,097,152 B
  float*  bias_p = (float*)(ws + 5242880);     // 2048*4 B
  float*  Hall   = (float*)(ws + 5308416);     // 2304*512*4  = 4,718,592 B
  float*  HS     = (float*)(ws + 10027008);    // 256*512*4   = 524,288 B

  pack_emb_k <<<(NVOC * HP + 255) / 256, 256, 0, stream>>>(emb, emb_p);
  pack_w_k   <<<(G4P * HP + 255) / 256, 256, 0, stream>>>(Wih, Wih_p);
  pack_w_k   <<<(G4P * HP + 255) / 256, 256, 0, stream>>>(Whh, Whh_p);
  pack_bias_k<<<(G4P + 255) / 256, 256, 0, stream>>>(bih, bhh, bias_p);
  zero_f32_k <<<(NB * HP + 255) / 256, 256, 0, stream>>>(HS, NB * HP);

  lstm_wmma_k<<<NTOT / ROWS_PER_WG, 512, 0, stream>>>(goals, hyps, emb_p, Wih_p, Whh_p,
                                                      bias_p, Hall);

  seg_sum_k<<<NHYP, 256, 0, stream>>>(Hall, seg, HS);
  mlp_k<<<NB, 256, 0, stream>>>(Hall, HS, W1, b1, W2, b2, Wf, bfin, out);
}
